// EnhancedVariationalHierarchicalGAT_2413771620931
// MI455X (gfx1250) — matrix operand order repaired and checked by reading, not compiled, not run on hardware
//
#include <hip/hip_runtime.h>
#include <hip/hip_bf16.h>

// ---------------- problem constants (match reference) ----------------
constexpr int N_NODES = 50000;
constexpr int E_EDGES = 600000;
constexpr int IN_F    = 128;
constexpr int H_HEADS = 8;
constexpr int C_CH    = 32;
constexpr int HC      = H_HEADS * C_CH;   // 256
constexpr int LAT     = 32;
constexpr int OUT_F   = 64;
constexpr int NCOM    = 30;
constexpr int ET      = E_EDGES + N_NODES; // edges + self loops
constexpr float BN_EPS = 1e-5f;
constexpr float NEG_SLOPE = 0.2f;

typedef __attribute__((ext_vector_type(16))) _Float16 v16h;
typedef __attribute__((ext_vector_type(4)))  _Float16 v4h;
typedef __attribute__((ext_vector_type(8)))  float    v8f;

// =====================================================================
// Weight pre-pack: W[K,P] f32 row-major  ->  f16 WMMA B-fragment layout.
// Fragment f = (kt, nt) covers K rows [kt*32, kt*32+32) x cols [nt*16, nt*16+16).
// Lane holds col = nt*16 + lane%16; halves j-pair at k = kt*32 + (lane/16)*16 + 2j.
// One thread emits the 16 halves of one (fragment, lane).
// =====================================================================
__global__ void pack_w_frag(const float* __restrict__ W, _Float16* __restrict__ Wp,
                            int K, int P)
{
    int idx = blockIdx.x * 256 + threadIdx.x;
    int total = (K >> 5) * (P >> 4) * 32;          // fragments * lanes
    if (idx >= total) return;
    int lane = idx & 31;
    int f    = idx >> 5;
    int nt   = f % (P >> 4);
    int kt   = f / (P >> 4);
    int col  = nt * 16 + (lane & 15);
    int kbase = kt * 32 + ((lane >> 4) << 4);
    _Float16* dst = Wp + (size_t)idx * 16;
    #pragma unroll
    for (int j = 0; j < 8; ++j) {
        dst[2 * j]     = (_Float16)W[(size_t)(kbase + 2 * j)     * P + col];
        dst[2 * j + 1] = (_Float16)W[(size_t)(kbase + 2 * j + 1) * P + col];
    }
}

// =====================================================================
// WMMA GEMM:  Y[M,P] = X[M,K] (*) Wp (+ bias) (+ Y if accumulate)
// A: f32 -> f16 staged through LDS (vector loads/stores, tile-level guard).
// B: pre-packed f16 fragments read directly from global (L2-resident).
// Block: 256 threads (8 waves), tile 128(M) x 64(P), K step 32, K templated.
// =====================================================================
#define TILE_M 128
#define TILE_N 64
#define TILE_K 32
#define LDSTRIDE 36   // halves; rows 72B-aligned so 4-half segments are 8B-aligned

template<int K>
__global__ __launch_bounds__(256)
void gemm_f16wmma(const float* __restrict__ X, const _Float16* __restrict__ Wp,
                  const float* __restrict__ bias, float* __restrict__ Y,
                  int M, int P, int accumulate)
{
    __shared__ _Float16 As[TILE_M][LDSTRIDE];      // [row][k]

    const int tid    = threadIdx.x;
    const int lane   = tid & 31;
    const int wave   = tid >> 5;
    const int wm     = wave & 3;                   // 0..3 -> rows wm*32
    const int wn     = wave >> 2;                  // 0..1 -> cols wn*32
    const int blockM = blockIdx.x * TILE_M;
    const int blockN = blockIdx.y * TILE_N;
    const int laneHi = lane >> 4;
    const int lane15 = lane & 15;
    const bool fullTile = (blockM + TILE_M <= M);

    const int ntBase  = (blockN >> 4) + wn * 2;    // B fragment column-tile base
    const int ntCount = P >> 4;
    const v16h* Wp16  = (const v16h*)Wp;

    v8f zero = {};
    v8f acc[2][2];
    acc[0][0] = zero; acc[0][1] = zero; acc[1][0] = zero; acc[1][1] = zero;

    const int rA  = tid >> 3;                      // 0..31 (row within 32-row slab)
    const int seg = (tid & 7) << 2;                // 0,4,...,28 (k segment)

    #pragma unroll
    for (int kt = 0; kt < (K >> 5); ++kt) {
        const int k0 = kt * TILE_K;

        // ---- stage A tile: 128x32 f32 -> f16 via float4 loads / v4h stores ----
        if (fullTile) {
            #pragma unroll
            for (int i = 0; i < 4; ++i) {
                int r = rA + i * 32;
                const float4 v = *(const float4*)(X + (size_t)(blockM + r) * K + k0 + seg);
                v4h hv; hv[0] = (_Float16)v.x; hv[1] = (_Float16)v.y;
                        hv[2] = (_Float16)v.z; hv[3] = (_Float16)v.w;
                *(v4h*)&As[r][seg] = hv;
            }
            if (k0 + TILE_K < K)   // speculative prefetch of next A tile
                __builtin_prefetch(X + (size_t)(blockM + rA) * K + k0 + TILE_K + seg, 0, 0);
        } else {
            #pragma unroll
            for (int i = 0; i < 4; ++i) {
                int r = rA + i * 32;
                int gr = blockM + r;
                float4 v = make_float4(0.f, 0.f, 0.f, 0.f);
                if (gr < M) v = *(const float4*)(X + (size_t)gr * K + k0 + seg);
                v4h hv; hv[0] = (_Float16)v.x; hv[1] = (_Float16)v.y;
                        hv[2] = (_Float16)v.z; hv[3] = (_Float16)v.w;
                *(v4h*)&As[r][seg] = hv;
            }
        }
        __syncthreads();

        // ---- A fragments from LDS (contiguous k = 16*(lane/16) + s, same order as packed B) ----
        union Frag { v16h h; unsigned u[8]; };
        Frag af[2];
        #pragma unroll
        for (int s = 0; s < 2; ++s) {
            int row = wm * 32 + s * 16 + lane15;
            const unsigned* ap = (const unsigned*)&As[row][0];
            #pragma unroll
            for (int j = 0; j < 8; ++j) {
                int kk = (laneHi << 4) + (j << 1);
                af[s].u[j] = ap[kk >> 1];
            }
        }
        // ---- B fragments straight from packed global ----
        v16h bf[2];
        #pragma unroll
        for (int s = 0; s < 2; ++s)
            bf[s] = Wp16[((size_t)kt * ntCount + ntBase + s) * 32 + lane];

        #pragma unroll
        for (int mi = 0; mi < 2; ++mi)
            #pragma unroll
            for (int ni = 0; ni < 2; ++ni)
                acc[mi][ni] = __builtin_amdgcn_wmma_f32_16x16x32_f16(
                    false, af[mi].h, false, bf[ni],
                    (short)0, acc[mi][ni], false, false);
        __syncthreads();
    }

    // ---- store: C/D layout: lanes 0-15 -> M=v, lanes 16-31 -> M=8+v, N=lane%16 ----
    #pragma unroll
    for (int mi = 0; mi < 2; ++mi) {
        #pragma unroll
        for (int ni = 0; ni < 2; ++ni) {
            int colg = blockN + wn * 32 + ni * 16 + lane15;
            float bv = bias ? bias[colg] : 0.0f;
            #pragma unroll
            for (int v = 0; v < 8; ++v) {
                int rowg = blockM + wm * 32 + mi * 16 + laneHi * 8 + v;
                if (rowg < M) {
                    size_t o = (size_t)rowg * P + colg;
                    float r = acc[mi][ni][v] + bv;
                    if (accumulate) r += Y[o];
                    Y[o] = r;
                }
            }
        }
    }
}

// =====================================================================
// GAT attention pipeline
// =====================================================================
__global__ void attn_dots(const float* __restrict__ h,
                          const float* __restrict__ asrc, const float* __restrict__ adst,
                          float* __restrict__ as_, float* __restrict__ ad_)
{
    int idx = blockIdx.x * 256 + threadIdx.x;            // n*H + head
    if (idx >= N_NODES * H_HEADS) return;
    int n = idx / H_HEADS, hd = idx - n * H_HEADS;
    const float* hp = h + (size_t)n * HC + hd * C_CH;
    const float* sv = asrc + hd * C_CH;
    const float* dv = adst + hd * C_CH;
    float vs = 0.f, vd = 0.f;
    #pragma unroll
    for (int c = 0; c < C_CH; ++c) { vs += hp[c] * sv[c]; vd += hp[c] * dv[c]; }
    as_[idx] = vs; ad_[idx] = vd;
}

__device__ __forceinline__ void edge_ends(const int* __restrict__ ei, int e, int& src, int& dst) {
    if (e < E_EDGES) { src = ei[e]; dst = ei[E_EDGES + e]; }
    else             { src = dst = e - E_EDGES; }           // self loop
}

// pass 1: leaky-relu logit, store to p, atomic max per (dst,head).
// float max via monotone uint encoding (uint 0 < every encoded real -> identity).
__global__ void edge_logits_max(const int* __restrict__ ei,
                                const float* __restrict__ as_, const float* __restrict__ ad_,
                                float* __restrict__ p, unsigned* __restrict__ m)
{
    long idx = (long)blockIdx.x * 256 + threadIdx.x;
    if (idx >= (long)ET * H_HEADS) return;
    int e = (int)(idx / H_HEADS), hd = (int)(idx - (long)e * H_HEADS);
    int src, dst; edge_ends(ei, e, src, dst);
    float el = as_[src * H_HEADS + hd] + ad_[dst * H_HEADS + hd];
    el = (el > 0.f) ? el : el * NEG_SLOPE;
    p[idx] = el;
    unsigned u = __float_as_uint(el);
    u = (u & 0x80000000u) ? ~u : (u | 0x80000000u);
    atomicMax(&m[dst * H_HEADS + hd], u);
}

// pass 2: p = exp(el - max); atomic sum per (dst,head)
__global__ void edge_softmax_num(const int* __restrict__ ei,
                                 float* __restrict__ p, const unsigned* __restrict__ m,
                                 float* __restrict__ s_)
{
    long idx = (long)blockIdx.x * 256 + threadIdx.x;
    if (idx >= (long)ET * H_HEADS) return;
    int e = (int)(idx / H_HEADS), hd = (int)(idx - (long)e * H_HEADS);
    int src, dst; edge_ends(ei, e, src, dst);
    unsigned u = m[dst * H_HEADS + hd];
    float mv = (u & 0x80000000u) ? __uint_as_float(u ^ 0x80000000u) : __uint_as_float(~u);
    float pe = __expf(p[idx] - mv);
    p[idx] = pe;
    atomicAdd(&s_[dst * H_HEADS + hd], pe);
}

// pass 3: one wave per edge; lane = channel within head, loop heads.
__global__ __launch_bounds__(256)
void edge_aggregate(const int* __restrict__ ei, const float* __restrict__ hlin,
                    const float* __restrict__ p, const float* __restrict__ s_,
                    float* __restrict__ out)
{
    int lane = threadIdx.x & 31;
    long w = (long)blockIdx.x * 8 + (threadIdx.x >> 5);
    if (w >= ET) return;
    int e = (int)w;
    int src, dst; edge_ends(ei, e, src, dst);
    const float* hs = hlin + (size_t)src * HC;
    float* op = out + (size_t)dst * HC;
    #pragma unroll
    for (int hd = 0; hd < H_HEADS; ++hd) {
        float alpha = p[(size_t)e * H_HEADS + hd] / (s_[dst * H_HEADS + hd] + 1e-16f);
        atomicAdd(&op[hd * C_CH + lane], hs[hd * C_CH + lane] * alpha);
    }
}

// =====================================================================
// BatchNorm (biased var, like torch/jax ref) + ReLU
// =====================================================================
__global__ __launch_bounds__(256)
void bn_stats(const float* __restrict__ x, float* __restrict__ stats, int M)
{
    int c = threadIdx.x;                 // channel 0..255
    int r0 = blockIdx.x * 64;
    int r1 = r0 + 64; if (r1 > M) r1 = M;
    float s = 0.f, ss = 0.f;
    for (int r = r0; r < r1; ++r) {
        float v = x[(size_t)r * HC + c];
        s += v; ss += v * v;
    }
    atomicAdd(&stats[c], s);
    atomicAdd(&stats[HC + c], ss);
}

__global__ void bn_apply_relu(float* __restrict__ x, const float* __restrict__ stats,
                              const float* __restrict__ g, const float* __restrict__ b, int M)
{
    long idx = (long)blockIdx.x * 256 + threadIdx.x;
    if (idx >= (long)M * HC) return;
    int c = (int)(idx & (HC - 1));
    float invM = 1.0f / (float)M;
    float mu  = stats[c] * invM;
    float var = stats[HC + c] * invM - mu * mu;
    float y = (x[idx] - mu) * rsqrtf(var + BN_EPS) * g[c] + b[c];
    x[idx] = (y > 0.f) ? y : 0.f;
}

// =====================================================================
// Small dense heads + reparameterization + fills
// =====================================================================
__global__ void small_gemm(const float* __restrict__ X, const float* __restrict__ W,
                           const float* __restrict__ b, float* __restrict__ Y,
                           int M, int K, int P)
{
    long idx = (long)blockIdx.x * 256 + threadIdx.x;
    if (idx >= (long)M * P) return;
    int n = (int)(idx / P), pc = (int)(idx - (long)n * P);
    float acc = b[pc];
    const float* xr = X + (size_t)n * K;
    for (int k = 0; k < K; ++k) acc += xr[k] * W[(size_t)k * P + pc];
    Y[idx] = acc;
}

__global__ void reparam(const float* __restrict__ mu, const float* __restrict__ lv,
                        const float* __restrict__ eps, float* __restrict__ z, long n)
{
    long idx = (long)blockIdx.x * 256 + threadIdx.x;
    if (idx >= n) return;
    z[idx] = mu[idx] + eps[idx] * __expf(0.5f * lv[idx]);
}

__global__ void fill_f32(float* __restrict__ p, float v, long n) {
    long idx = (long)blockIdx.x * 256 + threadIdx.x;
    if (idx < n) p[idx] = v;
}
__global__ void fill_u32(unsigned* __restrict__ p, unsigned v, long n) {
    long idx = (long)blockIdx.x * 256 + threadIdx.x;
    if (idx < n) p[idx] = v;
}
// aggregation buffer starts at the GAT bias (bias is added pre-BN in the ref)
__global__ void bias_rows(float* __restrict__ out, const float* __restrict__ b, long n) {
    long idx = (long)blockIdx.x * 256 + threadIdx.x;
    if (idx < n) out[idx] = b[idx & (HC - 1)];
}

// =====================================================================
// Orchestration
// =====================================================================
static inline unsigned cdiv(long a, long b) { return (unsigned)((a + b - 1) / b); }

extern "C" void kernel_launch(void* const* d_in, const int* in_sizes, int n_in,
                              void* d_out, int out_size, void* d_ws, size_t ws_size,
                              hipStream_t stream)
{
    const float* x    = (const float*)d_in[0];
    const float* eps  = (const float*)d_in[1];
    const int*   ei   = (const int*)d_in[2];
    const float* gW[3]  = {(const float*)d_in[3], (const float*)d_in[9],  (const float*)d_in[15]};
    const float* gas[3] = {(const float*)d_in[4], (const float*)d_in[10], (const float*)d_in[16]};
    const float* gad[3] = {(const float*)d_in[5], (const float*)d_in[11], (const float*)d_in[17]};
    const float* gb[3]  = {(const float*)d_in[6], (const float*)d_in[12], (const float*)d_in[18]};
    const float* bng[3] = {(const float*)d_in[7], (const float*)d_in[13], (const float*)d_in[19]};
    const float* bnb[3] = {(const float*)d_in[8], (const float*)d_in[14], (const float*)d_in[20]};
    const float* resW = (const float*)d_in[21]; const float* resb = (const float*)d_in[22];
    const float* muW  = (const float*)d_in[23]; const float* mub  = (const float*)d_in[24];
    const float* lvW  = (const float*)d_in[25]; const float* lvb  = (const float*)d_in[26];
    const float* decW = (const float*)d_in[27]; const float* decb = (const float*)d_in[28];
    const float* auxW = (const float*)d_in[29]; const float* auxb = (const float*)d_in[30];

    float* out    = (float*)d_out;                      // tuple: (out, mu, logvar, aux) flat
    float* outDec = out;                                // [N,64]
    float* outMu  = out    + (size_t)N_NODES * OUT_F;   // [N,32]
    float* outLv  = outMu  + (size_t)N_NODES * LAT;     // [N,32]
    float* outAux = outLv  + (size_t)N_NODES * LAT;     // [N,30]

    float* ws = (float*)d_ws;
    size_t o = 0;
    float*    hlin  = ws + o; o += (size_t)N_NODES * HC;       // GAT linear output
    float*    actA  = ws + o; o += (size_t)N_NODES * HC;       // ping
    float*    actB  = ws + o; o += (size_t)N_NODES * HC;       // pong
    float*    a_s   = ws + o; o += (size_t)N_NODES * H_HEADS;
    float*    a_d   = ws + o; o += (size_t)N_NODES * H_HEADS;
    unsigned* mbuf  = (unsigned*)(ws + o); o += (size_t)N_NODES * H_HEADS;
    float*    sbuf  = ws + o; o += (size_t)N_NODES * H_HEADS;
    float*    pbuf  = ws + o; o += (size_t)ET * H_HEADS;
    float*    stats = ws + o; o += 2 * HC;
    float*    zbuf  = ws + o; o += (size_t)N_NODES * LAT;
    // packed f16 weights (K*P halves each; offsets stay 32B-aligned)
    _Float16* wpG[3];
    wpG[0] = (_Float16*)(ws + o); o += (size_t)IN_F * HC / 2;  // gat1: 128x256
    wpG[1] = (_Float16*)(ws + o); o += (size_t)HC   * HC / 2;  // gat2: 256x256
    wpG[2] = (_Float16*)(ws + o); o += (size_t)HC   * HC / 2;  // gat3: 256x256
    _Float16* wpR = (_Float16*)(ws + o); o += (size_t)IN_F * HC / 2;   // res: 128x256

    // ---- pack all WMMA weights once (tiny) ----
    pack_w_frag<<<cdiv((long)IN_F * HC / 16, 256), 256, 0, stream>>>(gW[0], wpG[0], IN_F, HC);
    pack_w_frag<<<cdiv((long)HC   * HC / 16, 256), 256, 0, stream>>>(gW[1], wpG[1], HC,   HC);
    pack_w_frag<<<cdiv((long)HC   * HC / 16, 256), 256, 0, stream>>>(gW[2], wpG[2], HC,   HC);
    pack_w_frag<<<cdiv((long)IN_F * HC / 16, 256), 256, 0, stream>>>(resW,  wpR,    IN_F, HC);

    dim3 gemmGrid(cdiv(N_NODES, TILE_M), HC / TILE_N);  // 391 x 4

    float* acts[2] = {actA, actB};
    const float* cur = x;
    int curK = IN_F;

    for (int L = 0; L < 3; ++L) {
        float* aggr = acts[L & 1];   // L0->A, L1->B, L2->A (no aliasing with input)

        if (curK == IN_F)
            gemm_f16wmma<IN_F><<<gemmGrid, 256, 0, stream>>>(cur, wpG[L], nullptr, hlin,
                                                             N_NODES, HC, 0);
        else
            gemm_f16wmma<HC><<<gemmGrid, 256, 0, stream>>>(cur, wpG[L], nullptr, hlin,
                                                           N_NODES, HC, 0);

        attn_dots<<<cdiv((long)N_NODES * H_HEADS, 256), 256, 0, stream>>>(
            hlin, gas[L], gad[L], a_s, a_d);

        fill_u32<<<cdiv((long)N_NODES * H_HEADS, 256), 256, 0, stream>>>(
            mbuf, 0u, (long)N_NODES * H_HEADS);
        fill_f32<<<cdiv((long)N_NODES * H_HEADS, 256), 256, 0, stream>>>(
            sbuf, 0.f, (long)N_NODES * H_HEADS);
        bias_rows<<<cdiv((long)N_NODES * HC, 256), 256, 0, stream>>>(
            aggr, gb[L], (long)N_NODES * HC);

        edge_logits_max<<<cdiv((long)ET * H_HEADS, 256), 256, 0, stream>>>(
            ei, a_s, a_d, pbuf, mbuf);
        edge_softmax_num<<<cdiv((long)ET * H_HEADS, 256), 256, 0, stream>>>(
            ei, pbuf, mbuf, sbuf);
        edge_aggregate<<<cdiv(ET, 8), 256, 0, stream>>>(ei, hlin, pbuf, sbuf, aggr);

        fill_f32<<<cdiv(2 * HC, 256), 256, 0, stream>>>(stats, 0.f, 2 * HC);
        bn_stats<<<cdiv(N_NODES, 64), 256, 0, stream>>>(aggr, stats, N_NODES);
        bn_apply_relu<<<cdiv((long)N_NODES * HC, 256), 256, 0, stream>>>(
            aggr, stats, bng[L], bnb[L], N_NODES);

        cur = aggr;
        curK = HC;
    }

    float* hfin = acts[0];  // layer 3 output lives in actA

    // residual: hfin += x @ res_W + res_b  (accumulate-mode WMMA GEMM)
    gemm_f16wmma<IN_F><<<gemmGrid, 256, 0, stream>>>(x, wpR, resb, hfin,
                                                     N_NODES, HC, 1);

    // heads
    small_gemm<<<cdiv((long)N_NODES * LAT, 256), 256, 0, stream>>>(
        hfin, muW, mub, outMu, N_NODES, HC, LAT);
    small_gemm<<<cdiv((long)N_NODES * LAT, 256), 256, 0, stream>>>(
        hfin, lvW, lvb, outLv, N_NODES, HC, LAT);
    small_gemm<<<cdiv((long)N_NODES * NCOM, 256), 256, 0, stream>>>(
        hfin, auxW, auxb, outAux, N_NODES, HC, NCOM);

    reparam<<<cdiv((long)N_NODES * LAT, 256), 256, 0, stream>>>(
        outMu, outLv, eps, zbuf, (long)N_NODES * LAT);

    small_gemm<<<cdiv((long)N_NODES * OUT_F, 256), 256, 0, stream>>>(
        zbuf, decW, decb, outDec, N_NODES, LAT, OUT_F);
}
